// ConditionalTransformer_21603685499076
// MI455X (gfx1250) — matrix operand order repaired and checked by reading, not compiled, not run on hardware
//
#include <hip/hip_runtime.h>
#include <hip/hip_bf16.h>
#include <math.h>

// ---------------------------------------------------------------------------
// Model dims
// ---------------------------------------------------------------------------
#define BB        32
#define DSET      512
#define DOUTD     16
#define DHD       1024
#define HH        16
#define LL        6
#define MM        127
#define CC        8
#define SS        128          // M+1
#define DHEADD    64
#define DTOKD     33           // 1 + 2*DOUT
#define DIND      545          // DSET + DTOK
#define KEMB      576          // DIN padded to mult of 32
#define NTOK      4096         // B*S
#define NTOKH     4064         // B*M (head tokens)

// WMMA types ----------------------------------------------------------------
typedef __attribute__((ext_vector_type(16))) __bf16 v16bf;
typedef __attribute__((ext_vector_type(8)))  float  v8f;

struct alignas(16) U4 { unsigned int x, y, z, w; };
union Frag {
  v16bf bf;
  U4 q[2];
  unsigned short us[16];
};

__device__ __forceinline__ unsigned short f2bf(float f) {
  unsigned int u = __float_as_uint(f);
  unsigned int r = u + 0x7fffu + ((u >> 16) & 1u);   // round-to-nearest-even
  return (unsigned short)(r >> 16);
}

// Async copy of 32 contiguous bytes global -> LDS (per lane), no VGPR bounce.
// INST_OFFSET is added to BOTH the LDS and global addresses per CDNA5 ISA, so
// one base covers both 16B chunks.
__device__ __forceinline__ void async_copy32(unsigned ldsoff, const void* g) {
  asm volatile(
      "global_load_async_to_lds_b128 %0, %1, off\n\t"
      "global_load_async_to_lds_b128 %0, %1, off offset:16"
      :
      : "v"(ldsoff), "v"((unsigned long long)(uintptr_t)g)
      : "memory");
}

__device__ __forceinline__ void wait_async0() {
  asm volatile("s_wait_asynccnt 0x0" ::: "memory");
}

// ---------------------------------------------------------------------------
// Generic bf16 WMMA GEMM:  C[M,N] = A[M,K] * Bw[N,K]^T + bias[N]
// Block: 256 threads (8 waves). Tile: 128(M) x 128(N) x 32(K).
// Double-buffered LDS filled by GLOBAL_LOAD_ASYNC_TO_LDS_B128; compute on
// tile i overlaps the async fill of tile i+1. One barrier per K-step.
// M = gridDim.y*128, N = gridDim.x*128 (buffers padded, no edge guards).
// ---------------------------------------------------------------------------
__global__ __launch_bounds__(256)
void k_gemm(const unsigned short* __restrict__ A,
            const unsigned short* __restrict__ Bw,
            const float* __restrict__ bias,
            float* __restrict__ C,
            int K, int lda, int ldb, int ldc)
{
  __shared__ unsigned short As[2][128][40];  // 32 halfs + pad per row
  __shared__ unsigned short Bs[2][128][40];

  const int tid  = threadIdx.x;
  const int lane = tid & 31;
  const int wave = tid >> 5;
  const int bm   = blockIdx.y * 128;
  const int bn   = blockIdx.x * 128;

  v8f acc[8];
#pragma unroll
  for (int j = 0; j < 8; ++j)
#pragma unroll
    for (int e = 0; e < 8; ++e) acc[j][e] = 0.0f;

  const int lr = tid >> 1;            // 0..127 (row this thread stages)
  const int lc = (tid & 1) * 16;      // 0 or 16 (half offset)

  const unsigned short* gA = A + (size_t)(bm + lr) * lda + lc;
  const unsigned short* gB = Bw + (size_t)(bn + lr) * ldb + lc;
  const unsigned ldsA0 = (unsigned)(uintptr_t)&As[0][lr][lc];
  const unsigned ldsA1 = (unsigned)(uintptr_t)&As[1][lr][lc];
  const unsigned ldsB0 = (unsigned)(uintptr_t)&Bs[0][lr][lc];
  const unsigned ldsB1 = (unsigned)(uintptr_t)&Bs[1][lr][lc];

  const int mrow  = (wave << 4) + (lane & 15);
  const int abase = (lane < 16) ? 0 : 8;    // A frag K base within 32-chunk
  const int nlane = lane & 15;
  const int bbase = (lane < 16) ? 0 : 8;    // B frag K base within 32-chunk

  // prologue: stage tile 0 into buffer 0
  async_copy32(ldsA0, gA);
  async_copy32(ldsB0, gB);

  const int nk = K >> 5;
  for (int i = 0; i < nk; ++i) {
    const int cur = i & 1;
    wait_async0();        // this wave's async writes to buf 'cur' landed
    __syncthreads();      // everyone's writes visible; prev reads retired

    if (i + 1 < nk) {     // overlap next tile's fill with this tile's math
      async_copy32(cur ? ldsA0 : ldsA1, gA + (size_t)(i + 1) * 32);
      async_copy32(cur ? ldsB0 : ldsB1, gB + (size_t)(i + 1) * 32);
    }

    Frag af;
    af.q[0] = *(const U4*)&As[cur][mrow][abase];        // K = abase + 0..7
    af.q[1] = *(const U4*)&As[cur][mrow][abase + 16];   // K = abase + 16..23
    Frag bf[8];
#pragma unroll
    for (int j = 0; j < 8; ++j) {
      bf[j].q[0] = *(const U4*)&Bs[cur][j * 16 + nlane][bbase];
      bf[j].q[1] = *(const U4*)&Bs[cur][j * 16 + nlane][bbase + 16];
    }
#pragma unroll
    for (int j = 0; j < 8; ++j)
      acc[j] = __builtin_amdgcn_wmma_f32_16x16x32_bf16(
          false, af.bf, false, bf[j].bf, (short)0, acc[j], false, false);
  }

  const int roff = (lane < 16) ? 0 : 8;
#pragma unroll
  for (int j = 0; j < 8; ++j) {
    const int gn = bn + j * 16 + nlane;
    const float bv = bias ? bias[gn] : 0.0f;
#pragma unroll
    for (int r = 0; r < 8; ++r) {
      const int gm = bm + (wave << 4) + r + roff;
      C[(size_t)gm * ldc + gn] = acc[j][r] + bv;
    }
  }
}

// ---------------------------------------------------------------------------
// Fused attention: one block per (b,h). 256 threads = 8 waves; wave w owns
// query rows [w*16, w*16+16).
// Dynamic LDS: Sc (128x132 f32) + P (128x136 bf16) + Vt (64x136 bf16, V^T).
// ---------------------------------------------------------------------------
#define ATTN_SROW 132
#define ATTN_PROW 136
#define ATTN_SMEM (128 * ATTN_SROW * 4 + 128 * ATTN_PROW * 2 + 64 * ATTN_PROW * 2)

__global__ __launch_bounds__(256)
void k_attn(const unsigned short* __restrict__ qkvb,
            float* __restrict__ attn)
{
  extern __shared__ char smem[];
  float*          Sc = (float*)smem;
  unsigned short* P  = (unsigned short*)(smem + 128 * ATTN_SROW * 4);
  unsigned short* Vt = P + 128 * ATTN_PROW;

  const int b    = blockIdx.x >> 4;
  const int h    = blockIdx.x & 15;
  const int lane = threadIdx.x & 31;
  const int wave = threadIdx.x >> 5;

  const unsigned short* qbase = qkvb + (size_t)b * 128 * 3072 + h * 64;
  const unsigned short* kbase = qbase + 1024;
  const unsigned short* vbase = qbase + 2048;

  const int mloc  = lane & 15;
  const int qrow  = wave * 16 + mloc;
  const int abase = (lane < 16) ? 0 : 8;
  const int nlane = lane & 15;
  const int bbase = (lane < 16) ? 0 : 8;
  const int roff  = (lane < 16) ? 0 : 8;

  // ---- stage V transposed into LDS (coalesced reads) ----
  for (int idx = threadIdx.x; idx < 128 * 64; idx += 256) {
    const int k = idx >> 6, d = idx & 63;
    Vt[d * ATTN_PROW + k] = vbase[(size_t)k * 3072 + d];
  }

  // ---- scores = Q K^T / 8, causal mask ----
  Frag qf[2];
#pragma unroll
  for (int kk = 0; kk < 2; ++kk) {
    const unsigned short* qr = qbase + (size_t)qrow * 3072 + kk * 32;
    qf[kk].q[0] = *(const U4*)(qr + abase);
    qf[kk].q[1] = *(const U4*)(qr + abase + 16);
  }

  for (int kt = 0; kt < 8; ++kt) {
    Frag kf[2];
#pragma unroll
    for (int kk = 0; kk < 2; ++kk) {
      const unsigned short* kr =
          kbase + (size_t)(kt * 16 + nlane) * 3072 + kk * 32 + bbase;
      kf[kk].q[0] = *(const U4*)kr;
      kf[kk].q[1] = *(const U4*)(kr + 16);
    }
    v8f acc;
#pragma unroll
    for (int e = 0; e < 8; ++e) acc[e] = 0.0f;
#pragma unroll
    for (int kk = 0; kk < 2; ++kk)
      acc = __builtin_amdgcn_wmma_f32_16x16x32_bf16(
          false, qf[kk].bf, false, kf[kk].bf, (short)0, acc, false, false);
#pragma unroll
    for (int r = 0; r < 8; ++r) {
      const int qg = wave * 16 + r + roff;
      const int kg = kt * 16 + nlane;
      float v = acc[r] * 0.125f;            // 1/sqrt(64)
      if (kg > qg) v = -1e30f;
      Sc[qg * ATTN_SROW + kg] = v;
    }
  }
  __syncthreads();

  // ---- row softmax -> bf16 P ----
  if (threadIdx.x < 128) {
    float* row = Sc + threadIdx.x * ATTN_SROW;
    float mx = -3.4e38f;
    for (int j = 0; j < 128; ++j) mx = fmaxf(mx, row[j]);
    float sum = 0.0f;
    for (int j = 0; j < 128; ++j) { float e = __expf(row[j] - mx); row[j] = e; sum += e; }
    const float inv = 1.0f / sum;
    unsigned short* prow = P + threadIdx.x * ATTN_PROW;
    for (int j = 0; j < 128; ++j) prow[j] = f2bf(row[j] * inv);
  }
  __syncthreads();

  // ---- O = P V ----
  for (int nt = 0; nt < 4; ++nt) {
    v8f acc;
#pragma unroll
    for (int e = 0; e < 8; ++e) acc[e] = 0.0f;
    const int d = nt * 16 + nlane;
    const unsigned short* vr = Vt + d * ATTN_PROW;
#pragma unroll
    for (int ks = 0; ks < 4; ++ks) {
      Frag pf;
      const unsigned short* pr = P + qrow * ATTN_PROW + ks * 32;
      pf.q[0] = *(const U4*)(pr + abase);
      pf.q[1] = *(const U4*)(pr + abase + 16);
      Frag vf;
      vf.q[0] = *(const U4*)(vr + ks * 32 + bbase);
      vf.q[1] = *(const U4*)(vr + ks * 32 + bbase + 16);
      acc = __builtin_amdgcn_wmma_f32_16x16x32_bf16(
          false, pf.bf, false, vf.bf, (short)0, acc, false, false);
    }
#pragma unroll
    for (int r = 0; r < 8; ++r) {
      const int qg = wave * 16 + r + roff;
      attn[(size_t)(b * 128 + qg) * 1024 + h * 64 + d] = acc[r];
    }
  }
}

// ---------------------------------------------------------------------------
// Elementwise helpers
// ---------------------------------------------------------------------------
__global__ void k_f2b(const float* __restrict__ src,
                      unsigned short* __restrict__ dst, int n)
{
  for (int i = blockIdx.x * blockDim.x + threadIdx.x; i < n;
       i += gridDim.x * blockDim.x)
    dst[i] = f2bf(src[i]);
}

__global__ void k_relu_b(const float* __restrict__ src,
                         unsigned short* __restrict__ dst, int n)
{
  for (int i = blockIdx.x * blockDim.x + threadIdx.x; i < n;
       i += gridDim.x * blockDim.x)
    dst[i] = f2bf(fmaxf(src[i], 0.0f));
}

// W_emb (1024 x 545) -> bf16 padded (1024 x 576)
__global__ void k_pad_wemb(const float* __restrict__ W,
                           unsigned short* __restrict__ Wb)
{
  const int n = DHD * KEMB;
  for (int i = blockIdx.x * blockDim.x + threadIdx.x; i < n;
       i += gridDim.x * blockDim.x) {
    const int r = i / KEMB, c = i % KEMB;
    Wb[i] = (c < DIND) ? f2bf(W[r * DIND + c]) : (unsigned short)0;
  }
}

// Assemble embedding input A (4096 x 576) bf16: [cond | sos/targets | 0-pad]
__global__ void k_build_A(const float* __restrict__ input,
                          const float* __restrict__ targets,
                          const float* __restrict__ sos,
                          unsigned short* __restrict__ A)
{
  const int n = NTOK * KEMB;
  for (int i = blockIdx.x * blockDim.x + threadIdx.x; i < n;
       i += gridDim.x * blockDim.x) {
    const int t = i / KEMB, c = i % KEMB;
    const int b = t >> 7, s = t & 127;
    float v = 0.0f;
    if (c < DSET) v = input[b * DSET + c];
    else if (c < DIND) {
      const int j = c - DSET;
      v = (s == 0) ? sos[j] : targets[((size_t)b * MM + (s - 1)) * DTOKD + j];
    }
    A[i] = f2bf(v);
  }
}

// X += positional encoding, also emit bf16 copy
__global__ void k_add_pe(float* __restrict__ X, unsigned short* __restrict__ Xb)
{
  const int n = NTOK * DHD;
  const float kdiv = 9.210340371976184f / 1024.0f;  // ln(10000)/DH
  for (int i = blockIdx.x * blockDim.x + threadIdx.x; i < n;
       i += gridDim.x * blockDim.x) {
    const int t = i >> 10, d = i & 1023;
    const int s = t & 127;
    const int pairi = d >> 1;
    const float ang = (float)s * __expf(-(float)(2 * pairi) * kdiv);
    const float pe = (d & 1) ? __cosf(ang) : __sinf(ang);
    const float v = X[i] + pe;
    X[i] = v;
    Xb[i] = f2bf(v);
  }
}

// X = LayerNorm(X + Y) * g + be ; also emit bf16 copy
__global__ __launch_bounds__(256)
void k_res_ln(float* __restrict__ X, const float* __restrict__ Y,
              const float* __restrict__ g, const float* __restrict__ be,
              unsigned short* __restrict__ Xb)
{
  __shared__ float red[256];
  const size_t t = blockIdx.x;
  const int tid = threadIdx.x;
  float v[4];
  float s = 0.0f;
#pragma unroll
  for (int i = 0; i < 4; ++i) {
    const int d = tid + i * 256;
    const float x = X[t * 1024 + d] + Y[t * 1024 + d];
    v[i] = x; s += x;
  }
  red[tid] = s; __syncthreads();
  for (int o = 128; o > 0; o >>= 1) {
    if (tid < o) red[tid] += red[tid + o];
    __syncthreads();
  }
  const float mean = red[0] * (1.0f / 1024.0f);
  __syncthreads();
  float s2 = 0.0f;
#pragma unroll
  for (int i = 0; i < 4; ++i) { const float dd = v[i] - mean; s2 += dd * dd; }
  red[tid] = s2; __syncthreads();
  for (int o = 128; o > 0; o >>= 1) {
    if (tid < o) red[tid] += red[tid + o];
    __syncthreads();
  }
  const float rstd = rsqrtf(red[0] * (1.0f / 1024.0f) + 1e-5f);
#pragma unroll
  for (int i = 0; i < 4; ++i) {
    const int d = tid + i * 256;
    const float o = (v[i] - mean) * rstd * g[d] + be[d];
    X[t * 1024 + d] = o;
    Xb[t * 1024 + d] = f2bf(o);
  }
}

// Compact head tokens: Hb[t<4064] = Xb[row(b,m)], zero-pad rows 4064..4095
__global__ void k_compact(const unsigned short* __restrict__ Xb,
                          unsigned short* __restrict__ Hb)
{
  const int n = NTOK * DHD;
  for (int i = blockIdx.x * blockDim.x + threadIdx.x; i < n;
       i += gridDim.x * blockDim.x) {
    const int t = i >> 10, d = i & 1023;
    if (t < NTOKH) {
      const int b = t / MM, m = t % MM;
      Hb[i] = Xb[(size_t)(b * 128 + m) * 1024 + d];
    } else Hb[i] = 0;
  }
}

// masks + pi_m + pi_v heads (17 outputs/token); one wave per token
__global__ __launch_bounds__(32)
void k_small_heads(const float* __restrict__ X,
                   const float* __restrict__ Wm,  const float* __restrict__ bm,
                   const float* __restrict__ Wpm, const float* __restrict__ bpm,
                   const float* __restrict__ Wpv, const float* __restrict__ bpv,
                   float* __restrict__ out,
                   size_t offPiM, size_t offPiV, size_t offMask)
{
  const int t = blockIdx.x;
  const int lane = threadIdx.x;
  const int b = t / MM, m = t % MM;
  const float* hrow = X + (size_t)(b * 128 + m) * 1024;
  float dots[17];
#pragma unroll 1
  for (int o = 0; o < 17; ++o) {
    const float* w = (o == 0) ? Wm
                   : (o < 9)  ? (Wpm + (size_t)(o - 1) * 1024)
                              : (Wpv + (size_t)(o - 9) * 1024);
    float s = 0.0f;
    for (int k = lane; k < 1024; k += 32) s += hrow[k] * w[k];
    for (int off = 16; off > 0; off >>= 1) s += __shfl_down(s, off, 32);
    dots[o] = s;
  }
  if (lane == 0) {
    out[offMask + t] = dots[0] + bm[0];
    float e[8], mx, sum;
    mx = -3.4e38f;
    for (int j = 0; j < 8; ++j) { e[j] = dots[1 + j] + bpm[j]; mx = fmaxf(mx, e[j]); }
    sum = 0.0f;
    for (int j = 0; j < 8; ++j) { e[j] = __expf(e[j] - mx); sum += e[j]; }
    for (int j = 0; j < 8; ++j) out[offPiM + (size_t)t * 8 + j] = e[j] / sum;
    mx = -3.4e38f;
    for (int j = 0; j < 8; ++j) { e[j] = dots[9 + j] + bpv[j]; mx = fmaxf(mx, e[j]); }
    sum = 0.0f;
    for (int j = 0; j < 8; ++j) { e[j] = __expf(e[j] - mx); sum += e[j]; }
    for (int j = 0; j < 8; ++j) out[offPiV + (size_t)t * 8 + j] = e[j] / sum;
  }
}

// mu / var scatter: out[off + i] = (exp?) C[i], i over 4064*128
__global__ void k_head_out(const float* __restrict__ Csrc,
                           float* __restrict__ dst, int doExp)
{
  const int n = NTOKH * 128;
  for (int i = blockIdx.x * blockDim.x + threadIdx.x; i < n;
       i += gridDim.x * blockDim.x) {
    const float v = Csrc[i];
    dst[i] = doExp ? __expf(v) : v;
  }
}

// ---------------------------------------------------------------------------
// Host launcher
// ---------------------------------------------------------------------------
extern "C" void kernel_launch(void* const* d_in, const int* in_sizes, int n_in,
                              void* d_out, int out_size, void* d_ws, size_t ws_size,
                              hipStream_t stream)
{
  (void)in_sizes; (void)n_in; (void)out_size; (void)ws_size;
  const float* input   = (const float*)d_in[0];
  const float* targets = (const float*)d_in[1];
  const float* W_emb   = (const float*)d_in[2];
  const float* b_emb   = (const float*)d_in[3];
  const float* sos     = (const float*)d_in[4];
  const float* Wqkv    = (const float*)d_in[5];
  const float* bqkv    = (const float*)d_in[6];
  const float* Wo      = (const float*)d_in[7];
  const float* bo      = (const float*)d_in[8];
  const float* W1      = (const float*)d_in[9];
  const float* b1      = (const float*)d_in[10];
  const float* W2      = (const float*)d_in[11];
  const float* b2      = (const float*)d_in[12];
  const float* g1      = (const float*)d_in[13];
  const float* be1     = (const float*)d_in[14];
  const float* g2      = (const float*)d_in[15];
  const float* be2     = (const float*)d_in[16];
  const float* Wm      = (const float*)d_in[17];
  const float* bm      = (const float*)d_in[18];
  const float* Wpi_m   = (const float*)d_in[19];
  const float* bpi_m   = (const float*)d_in[20];
  const float* Wmu_m   = (const float*)d_in[21];
  const float* bmu_m   = (const float*)d_in[22];
  const float* Wsg_m   = (const float*)d_in[23];
  const float* bsg_m   = (const float*)d_in[24];
  const float* Wpi_v   = (const float*)d_in[25];
  const float* bpi_v   = (const float*)d_in[26];
  const float* Wmu_v   = (const float*)d_in[27];
  const float* bmu_v   = (const float*)d_in[28];
  const float* Wsg_v   = (const float*)d_in[29];
  const float* bsg_v   = (const float*)d_in[30];
  float* out = (float*)d_out;

  // output layout (flat, reference return order)
  const size_t OFF_PI_M  = 0;
  const size_t OFF_MU_M  = OFF_PI_M  + (size_t)NTOKH * CC;
  const size_t OFF_VAR_M = OFF_MU_M  + (size_t)NTOKH * 128;
  const size_t OFF_PI_V  = OFF_VAR_M + (size_t)NTOKH * 128;
  const size_t OFF_MU_V  = OFF_PI_V  + (size_t)NTOKH * CC;
  const size_t OFF_VAR_V = OFF_MU_V  + (size_t)NTOKH * 128;
  const size_t OFF_MASK  = OFF_VAR_V + (size_t)NTOKH * 128;

  // workspace carve-up
  char* ws = (char*)d_ws;
  size_t off = 0;
  auto take = [&](size_t bytes) -> char* {
    char* p = ws + off;
    off = (off + bytes + 255) & ~(size_t)255;
    return p;
  };
  const size_t SZ_WQKV = (size_t)LL * 3072 * 1024;
  const size_t SZ_WSQ  = (size_t)LL * 1024 * 1024;
  const size_t SZ_X    = (size_t)NTOK * DHD;
  const size_t SZ_QKV  = (size_t)NTOK * 3072;

  unsigned short* WqkvB = (unsigned short*)take(SZ_WQKV * 2);
  unsigned short* WoB   = (unsigned short*)take(SZ_WSQ * 2);
  unsigned short* W1B   = (unsigned short*)take(SZ_WSQ * 2);
  unsigned short* W2B   = (unsigned short*)take(SZ_WSQ * 2);
  unsigned short* WembB = (unsigned short*)take((size_t)DHD * KEMB * 2);
  unsigned short* WmuMB = (unsigned short*)take((size_t)128 * 1024 * 2);
  unsigned short* WsgMB = (unsigned short*)take((size_t)128 * 1024 * 2);
  unsigned short* WmuVB = (unsigned short*)take((size_t)128 * 1024 * 2);
  unsigned short* WsgVB = (unsigned short*)take((size_t)128 * 1024 * 2);
  unsigned short* Aemb  = (unsigned short*)take((size_t)NTOK * KEMB * 2);
  float*          X     = (float*)take(SZ_X * 4);
  unsigned short* Xb    = (unsigned short*)take(SZ_X * 2);
  float*          qkv   = (float*)take(SZ_QKV * 4);
  unsigned short* qkvb  = (unsigned short*)take(SZ_QKV * 2);
  float*          attn  = (float*)take(SZ_X * 4);
  unsigned short* attnb = (unsigned short*)take(SZ_X * 2);
  float*          y     = (float*)take(SZ_X * 4);
  unsigned short* ffb   = (unsigned short*)take(SZ_X * 2);
  unsigned short* Hb    = (unsigned short*)take(SZ_X * 2);
  float*          headC = (float*)take((size_t)NTOK * 128 * 4);

  const dim3 T256(256);

  // ---- weight conversion fp32 -> bf16 ----
  k_f2b<<<dim3(4096), T256, 0, stream>>>(Wqkv, WqkvB, (int)SZ_WQKV);
  k_f2b<<<dim3(2048), T256, 0, stream>>>(Wo,   WoB,   (int)SZ_WSQ);
  k_f2b<<<dim3(2048), T256, 0, stream>>>(W1,   W1B,   (int)SZ_WSQ);
  k_f2b<<<dim3(2048), T256, 0, stream>>>(W2,   W2B,   (int)SZ_WSQ);
  k_f2b<<<dim3(256),  T256, 0, stream>>>(Wmu_m, WmuMB, 128 * 1024);
  k_f2b<<<dim3(256),  T256, 0, stream>>>(Wsg_m, WsgMB, 128 * 1024);
  k_f2b<<<dim3(256),  T256, 0, stream>>>(Wmu_v, WmuVB, 128 * 1024);
  k_f2b<<<dim3(256),  T256, 0, stream>>>(Wsg_v, WsgVB, 128 * 1024);
  k_pad_wemb<<<dim3(1024), T256, 0, stream>>>(W_emb, WembB);

  // ---- embedding ----
  k_build_A<<<dim3(4096), T256, 0, stream>>>(input, targets, sos, Aemb);
  k_gemm<<<dim3(DHD / 128, NTOK / 128), T256, 0, stream>>>(
      Aemb, WembB, b_emb, X, KEMB, KEMB, KEMB, DHD);
  k_add_pe<<<dim3(4096), T256, 0, stream>>>(X, Xb);

  // ---- transformer layers ----
  for (int i = 0; i < LL; ++i) {
    const unsigned short* Wqb = WqkvB + (size_t)i * 3072 * 1024;
    const unsigned short* Wob = WoB  + (size_t)i * 1024 * 1024;
    const unsigned short* W1b = W1B  + (size_t)i * 1024 * 1024;
    const unsigned short* W2b = W2B  + (size_t)i * 1024 * 1024;

    k_gemm<<<dim3(3072 / 128, NTOK / 128), T256, 0, stream>>>(
        Xb, Wqb, bqkv + (size_t)i * 3072, qkv, 1024, 1024, 1024, 3072);
    k_f2b<<<dim3(4096), T256, 0, stream>>>(qkv, qkvb, (int)SZ_QKV);

    k_attn<<<dim3(BB * HH), T256, ATTN_SMEM, stream>>>(qkvb, attn);

    k_f2b<<<dim3(2048), T256, 0, stream>>>(attn, attnb, (int)SZ_X);
    k_gemm<<<dim3(1024 / 128, NTOK / 128), T256, 0, stream>>>(
        attnb, Wob, bo + (size_t)i * 1024, y, 1024, 1024, 1024, 1024);
    k_res_ln<<<dim3(NTOK), T256, 0, stream>>>(
        X, y, g1 + (size_t)i * 1024, be1 + (size_t)i * 1024, Xb);

    k_gemm<<<dim3(1024 / 128, NTOK / 128), T256, 0, stream>>>(
        Xb, W1b, b1 + (size_t)i * 1024, y, 1024, 1024, 1024, 1024);
    k_relu_b<<<dim3(2048), T256, 0, stream>>>(y, ffb, (int)SZ_X);
    k_gemm<<<dim3(1024 / 128, NTOK / 128), T256, 0, stream>>>(
        ffb, W2b, b2 + (size_t)i * 1024, y, 1024, 1024, 1024, 1024);
    k_res_ln<<<dim3(NTOK), T256, 0, stream>>>(
        X, y, g2 + (size_t)i * 1024, be2 + (size_t)i * 1024, Xb);
  }

  // ---- heads ----
  k_compact<<<dim3(2048), T256, 0, stream>>>(Xb, Hb);
  k_small_heads<<<dim3(NTOKH), dim3(32), 0, stream>>>(
      X, Wm, bm, Wpi_m, bpi_m, Wpi_v, bpi_v, out, OFF_PI_M, OFF_PI_V, OFF_MASK);

  k_gemm<<<dim3(1, NTOK / 128), T256, 0, stream>>>(
      Hb, WmuMB, bmu_m, headC, 1024, 1024, 1024, 128);
  k_head_out<<<dim3(512), T256, 0, stream>>>(headC, out + OFF_MU_M, 0);

  k_gemm<<<dim3(1, NTOK / 128), T256, 0, stream>>>(
      Hb, WsgMB, bsg_m, headC, 1024, 1024, 1024, 128);
  k_head_out<<<dim3(512), T256, 0, stream>>>(headC, out + OFF_VAR_M, 1);

  k_gemm<<<dim3(1, NTOK / 128), T256, 0, stream>>>(
      Hb, WmuVB, bmu_v, headC, 1024, 1024, 1024, 128);
  k_head_out<<<dim3(512), T256, 0, stream>>>(headC, out + OFF_MU_V, 0);

  k_gemm<<<dim3(1, NTOK / 128), T256, 0, stream>>>(
      Hb, WsgVB, bsg_v, headC, 1024, 1024, 1024, 128);
  k_head_out<<<dim3(512), T256, 0, stream>>>(headC, out + OFF_VAR_V, 1);
}
